// FlowAttention_51281909514577
// MI455X (gfx1250) — matrix operand order repaired and checked by reading, not compiled, not run on hardware
//
#include <hip/hip_runtime.h>
#include <hip/hip_bf16.h>

// FlowAttention for MI455X (gfx1250): bf16 WMMA flash-attention.
// B=4, C=256, H=W=64 -> N=4096 tokens per batch.

#define NTOK 4096
#define CDIM 256

typedef __attribute__((ext_vector_type(16))) __bf16 bf16x16;
typedef __attribute__((ext_vector_type(8)))  float  f32x8;

union FragU {
  bf16x16 v;
  uint4   q[2];
  __bf16  e[16];
};

// gfx1250 async global->LDS builtins (signature revealed by round-2 probe:
// (v4i addrspace(1)*, v4i addrspace(3)*, imm offset, imm cpol)).
#if defined(__HIP_DEVICE_COMPILE__)
#if __has_builtin(__builtin_amdgcn_global_load_async_to_lds_b128) && \
    __has_builtin(__builtin_amdgcn_s_wait_asynccnt)
#define USE_ASYNC_LDS 1
#endif
#endif

#if defined(USE_ASYNC_LDS)
typedef int v4i __attribute__((__vector_size__(4 * sizeof(int))));
typedef __attribute__((address_space(1))) v4i* v4i_g;
typedef __attribute__((address_space(3))) v4i* v4i_l;
__device__ __forceinline__ void async_cp16(const void* g, void* l) {
  __builtin_amdgcn_global_load_async_to_lds_b128((v4i_g)g, (v4i_l)l, 0, 0);
}
#endif

// Fast base-2 exp: raw v_exp_f32 when available.
__device__ __forceinline__ float fast_exp2(float x) {
#if __has_builtin(__builtin_amdgcn_exp2f)
  return __builtin_amdgcn_exp2f(x);
#else
  return exp2f(x);
#endif
}

// Reduce across the 16 lanes of each wave-half (rows 0-7 live in lanes 0-15,
// rows 8-15 in lanes 16-31 for WMMA C/D layout, so halves reduce independently).
__device__ __forceinline__ float rmax16(float v) {
#pragma unroll
  for (int m = 1; m < 16; m <<= 1) v = fmaxf(v, __shfl_xor(v, m, 32));
  return v;
}
__device__ __forceinline__ float rsum16(float v) {
#pragma unroll
  for (int m = 1; m < 16; m <<= 1) v += __shfl_xor(v, m, 32);
  return v;
}

// ---------------------------------------------------------------------------
// Kernel 0: convert f32 weights to bf16 (tiny; weights reused ~16k times).
// ---------------------------------------------------------------------------
__global__ __launch_bounds__(256)
void cw_kernel(const float* __restrict__ qw, const float* __restrict__ kw,
               __bf16* __restrict__ wqb, __bf16* __restrict__ wkb) {
  int i = blockIdx.x * 256 + threadIdx.x;           // 0 .. 131071
  if (i < CDIM * CDIM) {
    wqb[i] = (__bf16)qw[i];
  } else {
    int j = i - CDIM * CDIM;
    wkb[j] = (__bf16)kw[j];
  }
}

// ---------------------------------------------------------------------------
// Kernel 1: Q/K projection.  tok[n][c] = feature[b][c][n]; out = tok * W^T + b
// Block: 64 tokens staged+transposed to LDS in bf16.  Per k-step the block
// also stages a 32-wide slab of BOTH weight matrices into LDS so B-fragments
// are ds_load_b128 (no per-WMMA global-load latency).  8 waves: 0-3 -> Q,
// 4-7 -> K.  Output Q,K stored bf16 row-major [b][n][256].
// ---------------------------------------------------------------------------
__global__ __launch_bounds__(256)
void proj_kernel(const float* __restrict__ feature,
                 const float* __restrict__ q_b,
                 const float* __restrict__ k_b,
                 const __bf16* __restrict__ wqb,
                 const __bf16* __restrict__ wkb,
                 __bf16* __restrict__ Qb,
                 __bf16* __restrict__ Kb) {
  __shared__ __align__(16) __bf16 tokLds[64][CDIM];   // 32 KB
  __shared__ __align__(16) __bf16 wLds[2][CDIM][32];  // 32 KB (Q slab, K slab)

  const int tid = threadIdx.x;
  const int n0  = blockIdx.x * 64;
  const int b   = blockIdx.y;

  // Stage feature[b][c][n0..n0+64) -> tokLds[n][c] with f32->bf16 convert.
  {
    const float* fb = feature + (size_t)b * CDIM * NTOK + n0;
#pragma unroll
    for (int k = 0; k < 16; ++k) {
      int i  = tid + k * 256;              // 0..4095 (256 c-rows x 16 float4)
      int c  = i >> 4;
      int n4 = (i & 15) << 2;
      const float4 vv = *(const float4*)(fb + (size_t)c * NTOK + n4);
      tokLds[n4 + 0][c] = (__bf16)vv.x;
      tokLds[n4 + 1][c] = (__bf16)vv.y;
      tokLds[n4 + 2][c] = (__bf16)vv.z;
      tokLds[n4 + 3][c] = (__bf16)vv.w;
    }
  }

  const int w    = tid >> 5;
  const int lane = tid & 31;
  const int half = lane >> 4;
  const int l15  = lane & 15;
  const bool isQ = (w < 4);
  const int wrow = (w & 3) * 16;

  const float*  bias = isQ ? q_b : k_b;
  __bf16*       dst  = isQ ? Qb  : Kb;
  const int     mat  = isQ ? 0 : 1;

  f32x8 acc[16];
#pragma unroll
  for (int t = 0; t < 16; ++t) acc[t] = (f32x8){0, 0, 0, 0, 0, 0, 0, 0};

  for (int kc = 0; kc < 8; ++kc) {                    // K = 256 in steps of 32
    __syncthreads();                                  // wLds reuse / tok ready
    // Stage weight slab: wLds[m][d][0..32) = w_m[d][32*kc .. +32), m=Q,K.
    {
#pragma unroll
      for (int j = 0; j < 8; ++j) {
        int i   = tid + j * 256;                      // 0..2047 uint4 chunks
        int m   = i >> 10;                            // 1024 chunks per matrix
        int d   = (i >> 2) & 255;
        int qo  = (i & 3) * 8;                        // element offset in slab
        const __bf16* src = (m == 0 ? wqb : wkb) + (size_t)d * CDIM + 32 * kc + qo;
        *(uint4*)&wLds[m][d][qo] = *(const uint4*)src;
      }
    }
    __syncthreads();

    FragU af;                                          // A: tok[16 x 32]
    {
      const int row = wrow + l15;
      const int lo  = 32 * kc + (half ? 8 : 0);        // A striping
      af.q[0] = *(const uint4*)&tokLds[row][lo];
      af.q[1] = *(const uint4*)&tokLds[row][lo + 16];
    }
#pragma unroll
    for (int dt = 0; dt < 16; ++dt) {                  // 256 outputs / 16
      FragU bfr;                                       // B: W^T[32 x 16]
      const int d = dt * 16 + l15;                     // B column = W row
      const __bf16* p = &wLds[mat][d][half ? 8 : 0];   // B striping
      bfr.q[0] = *(const uint4*)(p);
      bfr.q[1] = *(const uint4*)(p + 16);
      acc[dt] = __builtin_amdgcn_wmma_f32_16x16x32_bf16(
          false, af.v, false, bfr.v, (short)0, acc[dt], false, false);
    }
  }

  // Bias + store bf16 (C/D layout: lane holds column d, rows striped in VGPRs).
#pragma unroll
  for (int dt = 0; dt < 16; ++dt) {
    const int d  = dt * 16 + l15;
    const float bv = bias[d];
#pragma unroll
    for (int r = 0; r < 8; ++r) {
      const int row = wrow + (half ? 8 + r : r);
      dst[((size_t)b * NTOK + n0 + row) * CDIM + d] = (__bf16)(acc[dt][r] + bv);
    }
  }
}

// ---------------------------------------------------------------------------
// Kernel 2: flash attention, 64-key tiles, double-buffered K staging.
// Each wave owns 16 query rows (Q fragments resident in 64 VGPRs).  The block
// stages each 64-key x 256 K-tile into LDS (async global->LDS if available,
// else pipelined global->VGPR->ds_store overlapped with compute), computes
// S = Q K^T / 16 with 32 WMMAs, online softmax (cross-lane reductions per
// wave-half, base-2 exp), restripes P via per-wave LDS into A-fragment
// layout, and accumulates out += P * V with 2 WMMAs (V padded 2->16 cols).
// ---------------------------------------------------------------------------
__global__ __launch_bounds__(256)
void flash_kernel(const __bf16* __restrict__ Qb,
                  const __bf16* __restrict__ Kb,
                  const float* __restrict__ flow,
                  float* __restrict__ out) {
  __shared__ __align__(16) __bf16 Kt[2][64][CDIM];    // 64 KB, double-buffered
  __shared__ __align__(16) __bf16 pLds[8][16][64];    // 16 KB, per-wave P

  const int tid  = threadIdx.x;
  const int w    = tid >> 5;
  const int lane = tid & 31;
  const int half = lane >> 4;
  const int l15  = lane & 15;
  const int b    = blockIdx.y;
  const int q0   = blockIdx.x * 128 + w * 16;

  const uint4* ksrc = (const uint4*)(Kb + (size_t)b * NTOK * CDIM);
  // tile s0: uint4 chunk base = s0*CDIM/8; per thread 8 chunks of 2048.

  // Resident Q fragments: rows q0..q0+15, full C=256 (8 fragments of K=32).
  FragU qf[8];
  {
    const __bf16* qrow = Qb + ((size_t)b * NTOK + q0 + l15) * CDIM;
#pragma unroll
    for (int kc = 0; kc < 8; ++kc) {
      const int lo = 32 * kc + (half ? 8 : 0);
      qf[kc].q[0] = *(const uint4*)(qrow + lo);
      qf[kc].q[1] = *(const uint4*)(qrow + lo + 16);
    }
  }

  // 1/sqrt(256) folded with log2(e): p = exp2(s*scale2 - m).
  const float scale2 = 0.0625f * 1.44269504f;
  f32x8 oacc = {0, 0, 0, 0, 0, 0, 0, 0};
  float mrun[8], lrun[8];
#pragma unroll
  for (int r = 0; r < 8; ++r) { mrun[r] = -3.0e38f; lrun[r] = 0.0f; }

  const int vc = l15;                                 // V/B column for PV GEMM

  // ---- preload tile 0 into buffer 0 ----
#if defined(USE_ASYNC_LDS)
  {
    uint4* dst = (uint4*)&Kt[0][0][0];
#pragma unroll
    for (int j = 0; j < 8; ++j) {
      int i = tid + j * 256;
      async_cp16(ksrc + i, dst + i);
    }
  }
#else
  {
    uint4* dst = (uint4*)&Kt[0][0][0];
#pragma unroll
    for (int j = 0; j < 8; ++j) {
      int i = tid + j * 256;
      dst[i] = ksrc[i];
    }
  }
#endif

  int cur = 0;
  for (int s0 = 0; s0 < NTOK; s0 += 64) {
#if defined(USE_ASYNC_LDS)
    __builtin_amdgcn_s_wait_asynccnt(0);
#endif
    __syncthreads();                                  // cur valid, nxt free

#if defined(USE_ASYNC_LDS)
    if (s0 + 64 < NTOK) {                             // kick next tile (async)
      const uint4* src = ksrc + (size_t)(s0 + 64) * (CDIM / 8);
      uint4* dst = (uint4*)&Kt[cur ^ 1][0][0];
#pragma unroll
      for (int j = 0; j < 8; ++j) {
        int i = tid + j * 256;
        async_cp16(src + i, dst + i);
      }
    }
#else
    uint4 stg[8];                                     // pipeline: load now,
    if (s0 + 64 < NTOK) {                             // ds_store after compute
      const uint4* src = ksrc + (size_t)(s0 + 64) * (CDIM / 8);
#pragma unroll
      for (int j = 0; j < 8; ++j) stg[j] = src[tid + j * 256];
    }
    if (s0 + 128 < NTOK) {                            // gfx1250 prefetch path
      const __bf16* pfb = Kb + ((size_t)b * NTOK + s0 + 128) * CDIM;
      __builtin_prefetch(pfb + (size_t)tid * 32, 0, 0);
      __builtin_prefetch(pfb + (size_t)tid * 32 + 8192, 0, 0);
    }
#endif

    // ---- S = Q * K^T : four 16x16 score tiles over 64 keys ----
    f32x8 sa0 = {0, 0, 0, 0, 0, 0, 0, 0};
    f32x8 sa1 = {0, 0, 0, 0, 0, 0, 0, 0};
    f32x8 sa2 = {0, 0, 0, 0, 0, 0, 0, 0};
    f32x8 sa3 = {0, 0, 0, 0, 0, 0, 0, 0};
    for (int kc = 0; kc < 8; ++kc) {
      const int lo = 32 * kc + (half ? 8 : 0);        // B striping
      FragU k0f, k1f, k2f, k3f;                       // B = K^T[32 x 16]
      const __bf16* kr0 = &Kt[cur][l15][lo];
      const __bf16* kr1 = &Kt[cur][16 + l15][lo];
      const __bf16* kr2 = &Kt[cur][32 + l15][lo];
      const __bf16* kr3 = &Kt[cur][48 + l15][lo];
      k0f.q[0] = *(const uint4*)(kr0); k0f.q[1] = *(const uint4*)(kr0 + 16);
      k1f.q[0] = *(const uint4*)(kr1); k1f.q[1] = *(const uint4*)(kr1 + 16);
      k2f.q[0] = *(const uint4*)(kr2); k2f.q[1] = *(const uint4*)(kr2 + 16);
      k3f.q[0] = *(const uint4*)(kr3); k3f.q[1] = *(const uint4*)(kr3 + 16);
      sa0 = __builtin_amdgcn_wmma_f32_16x16x32_bf16(
          false, qf[kc].v, false, k0f.v, (short)0, sa0, false, false);
      sa1 = __builtin_amdgcn_wmma_f32_16x16x32_bf16(
          false, qf[kc].v, false, k1f.v, (short)0, sa1, false, false);
      sa2 = __builtin_amdgcn_wmma_f32_16x16x32_bf16(
          false, qf[kc].v, false, k2f.v, (short)0, sa2, false, false);
      sa3 = __builtin_amdgcn_wmma_f32_16x16x32_bf16(
          false, qf[kc].v, false, k3f.v, (short)0, sa3, false, false);
    }

    // ---- online softmax over 64 keys ----
#pragma unroll
    for (int r = 0; r < 8; ++r) {
      float x0 = sa0[r] * scale2;
      float x1 = sa1[r] * scale2;
      float x2 = sa2[r] * scale2;
      float x3 = sa3[r] * scale2;
      float mx   = rmax16(fmaxf(fmaxf(x0, x1), fmaxf(x2, x3)));
      float mnew = fmaxf(mrun[r], mx);
      float alpha = fast_exp2(mrun[r] - mnew);
      float p0 = fast_exp2(x0 - mnew);
      float p1 = fast_exp2(x1 - mnew);
      float p2 = fast_exp2(x2 - mnew);
      float p3 = fast_exp2(x3 - mnew);
      float rs = rsum16((p0 + p1) + (p2 + p3));
      lrun[r] = lrun[r] * alpha + rs;
      mrun[r] = mnew;
      oacc[r] *= alpha;
      const int row = half ? 8 + r : r;               // restripe P via LDS
      pLds[w][row][l15]      = (__bf16)p0;
      pLds[w][row][l15 + 16] = (__bf16)p1;
      pLds[w][row][l15 + 32] = (__bf16)p2;
      pLds[w][row][l15 + 48] = (__bf16)p3;
    }

    // ---- out += P * V over the 64 keys (two K=32 WMMAs) ----
#pragma unroll
    for (int fi = 0; fi < 2; ++fi) {
      FragU pf;                                       // A layout from LDS
      const int lo = 32 * fi + (half ? 16 : 0);
      pf.q[0] = *(const uint4*)&pLds[w][l15][lo];
      pf.q[1] = *(const uint4*)&pLds[w][l15][lo + 8];

      FragU vf;                                       // B: column vc, K = s
      if (vc < 2) {
        const float* vp = flow + ((size_t)b * 2 + vc) * NTOK + s0 + 32 * fi +
                          (half ? 16 : 0);
        const float4 a0 = *(const float4*)(vp);
        const float4 a1 = *(const float4*)(vp + 4);
        const float4 a2 = *(const float4*)(vp + 8);
        const float4 a3 = *(const float4*)(vp + 12);
        vf.e[0]  = (__bf16)a0.x; vf.e[1]  = (__bf16)a0.y;
        vf.e[2]  = (__bf16)a0.z; vf.e[3]  = (__bf16)a0.w;
        vf.e[4]  = (__bf16)a1.x; vf.e[5]  = (__bf16)a1.y;
        vf.e[6]  = (__bf16)a1.z; vf.e[7]  = (__bf16)a1.w;
        vf.e[8]  = (__bf16)a2.x; vf.e[9]  = (__bf16)a2.y;
        vf.e[10] = (__bf16)a2.z; vf.e[11] = (__bf16)a2.w;
        vf.e[12] = (__bf16)a3.x; vf.e[13] = (__bf16)a3.y;
        vf.e[14] = (__bf16)a3.z; vf.e[15] = (__bf16)a3.w;
      } else {
#pragma unroll
        for (int j = 0; j < 16; ++j) vf.e[j] = (__bf16)0.0f;
      }
      oacc = __builtin_amdgcn_wmma_f32_16x16x32_bf16(
          false, pf.v, false, vf.v, (short)0, oacc, false, false);
    }

#if !defined(USE_ASYNC_LDS)
    if (s0 + 64 < NTOK) {                             // publish next tile
      uint4* dst = (uint4*)&Kt[cur ^ 1][0][0];
#pragma unroll
      for (int j = 0; j < 8; ++j) dst[tid + j * 256] = stg[j];
    }
#endif
    cur ^= 1;
  }

  // Epilogue: out[b][vc][n] = oacc / l.  Only columns 0,1 are real.
  if (vc < 2) {
    float* op = out + ((size_t)b * 2 + vc) * NTOK + q0;
#pragma unroll
    for (int r = 0; r < 8; ++r) {
      const int row = half ? 8 + r : r;
      op[row] = oacc[r] / lrun[r];
    }
  }
}

// ---------------------------------------------------------------------------
extern "C" void kernel_launch(void* const* d_in, const int* in_sizes, int n_in,
                              void* d_out, int out_size, void* d_ws, size_t ws_size,
                              hipStream_t stream) {
  (void)in_sizes; (void)n_in; (void)out_size; (void)ws_size;

  const float* feature = (const float*)d_in[0];   // [4,256,64,64]
  const float* flow    = (const float*)d_in[1];   // [4,2,64,64]
  const float* q_w     = (const float*)d_in[2];   // [256,256]
  const float* q_b     = (const float*)d_in[3];   // [256]
  const float* k_w     = (const float*)d_in[4];   // [256,256]
  const float* k_b     = (const float*)d_in[5];   // [256]
  float* out           = (float*)d_out;           // [4,2,64,64]

  // Workspace layout (bytes): Qb 8MiB | Kb 8MiB | wqb 128KiB | wkb 128KiB
  char* ws = (char*)d_ws;
  __bf16* Qb  = (__bf16*)(ws);
  __bf16* Kb  = (__bf16*)(ws + (size_t)8 * 1024 * 1024);
  __bf16* wqb = (__bf16*)(ws + (size_t)16 * 1024 * 1024);
  __bf16* wkb = (__bf16*)(ws + (size_t)16 * 1024 * 1024 + 131072);

  cw_kernel<<<dim3(512), dim3(256), 0, stream>>>(q_w, k_w, wqb, wkb);
  proj_kernel<<<dim3(64, 4), dim3(256), 0, stream>>>(feature, q_b, k_b,
                                                     wqb, wkb, Qb, Kb);
  flash_kernel<<<dim3(32, 4), dim3(256), 0, stream>>>(Qb, Kb, flow, out);
}
